// DifferentialAttention_50723563766133
// MI455X (gfx1250) — matrix operand order repaired and checked by reading, not compiled, not run on hardware
//
#include <hip/hip_runtime.h>
#include <stdint.h>

#define DEVINL __device__ __forceinline__

typedef __attribute__((ext_vector_type(16))) __bf16 v16bf;
typedef __attribute__((ext_vector_type(8)))  __bf16 v8bf;
typedef __attribute__((ext_vector_type(4)))  __bf16 v4bf;
typedef __attribute__((ext_vector_type(8)))  float  v8f;
typedef __attribute__((ext_vector_type(4)))  float  v4f;
typedef __attribute__((ext_vector_type(4)))  unsigned int u32x4;
typedef __attribute__((ext_vector_type(8)))  int          i32x8;
typedef __attribute__((ext_vector_type(4)))  int          i32x4;

constexpr int Bc = 2, Hc = 12, Sc = 2048, Dc = 128;
constexpr int QT = 64;   // query rows per block (4 waves x 16)
constexpr int KT = 32;   // keys per flash iteration
// 1/sqrt(128) * log2(e): softmax runs in the exp2 domain
constexpr float SCALE_LOG2 = 0.088388347648318447f * 1.4426950408889634f;

#if __has_builtin(__builtin_amdgcn_exp2f)
#define EXP2F __builtin_amdgcn_exp2f   // raw v_exp_f32; args <= 0, FTZ is fine
#else
#define EXP2F exp2f
#endif

#if __has_builtin(__builtin_amdgcn_tensor_load_to_lds)
#define HAVE_TDM 1
#else
#define HAVE_TDM 0
#endif

DEVINL v8f wmma_bf16(v16bf a, v16bf b, v8f c) {
  return __builtin_amdgcn_wmma_f32_16x16x32_bf16(
      false, a, false, b, (short)0, c, false, false);
}

union AFrag { v16bf v; v8bf h[2]; };

// butterfly exchange within 16-lane halves (ds_swizzle group-of-32 xor)
template <int M>
DEVINL float swz_xor(float x) {
  return __int_as_float(
      __builtin_amdgcn_ds_swizzle(__float_as_int(x), (M << 10) | 0x1f));
}

DEVINL float half_max(float x) {
  x = fmaxf(x, swz_xor<1>(x));
  x = fmaxf(x, swz_xor<2>(x));
  x = fmaxf(x, swz_xor<4>(x));
  x = fmaxf(x, swz_xor<8>(x));
  return x;
}

DEVINL float half_sum(float x) {
  x += swz_xor<1>(x);
  x += swz_xor<2>(x);
  x += swz_xor<4>(x);
  x += swz_xor<8>(x);
  return x;
}

// ---------------- pre-pass kernels: fp32 -> bf16 into workspace ----------------

__global__ __launch_bounds__(256) void cvt_bf16_kernel(
    const float* __restrict__ src, __bf16* __restrict__ dst) {
  const size_t i = ((size_t)blockIdx.x * 256 + threadIdx.x) * 4;
  v4f a = *(const v4f*)(src + i);
  v4bf r;
  r[0] = (__bf16)a.x; r[1] = (__bf16)a.y; r[2] = (__bf16)a.z; r[3] = (__bf16)a.w;
  *(v4bf*)(dst + i) = r;
}

// V [bh][s][d] fp32 -> VT [bh][d][s] bf16, 32x32 LDS-tiled transpose
__global__ __launch_bounds__(256) void vT_bf16_kernel(
    const float* __restrict__ v, __bf16* __restrict__ vt) {
  __shared__ float tile[32][33];
  const int t  = threadIdx.x;
  const int c  = t & 31;
  const int r0 = t >> 5;
  const int dt = blockIdx.x & 3;            // D/32
  const int st = (blockIdx.x >> 2) & 63;    // S/32
  const int bh = blockIdx.x >> 8;           // B*H
  const size_t in_base  = ((size_t)bh * Sc + st * 32) * Dc + dt * 32;
  const size_t out_base = ((size_t)bh * Dc + dt * 32) * Sc + st * 32;
  #pragma unroll
  for (int k = 0; k < 4; ++k) {
    const int r = r0 + k * 8;
    tile[r][c] = v[in_base + (size_t)r * Dc + c];
  }
  __syncthreads();
  #pragma unroll
  for (int k = 0; k < 4; ++k) {
    const int r = r0 + k * 8;
    vt[out_base + (size_t)r * Sc + c] = (__bf16)tile[c][r];
  }
}

// ---------------- async / TDM copy helpers ----------------

// contiguous 8KB bf16 tile -> LDS (4 x b128 per thread), ASYNCcnt-tracked
DEVINL void async_tile_8k(const __bf16* __restrict__ gsrc, __bf16* ldst, int tid) {
  #pragma unroll
  for (int i = 0; i < 4; ++i) {
    const int e = (i * 128 + tid) * 8;   // 16B chunks
    asm volatile("global_load_async_to_lds_b128 %0, %1, off"
                 :: "v"((uint32_t)(uintptr_t)(ldst + e)),
                    "v"((uint64_t)(uintptr_t)(gsrc + e)) : "memory");
  }
}

#if HAVE_TDM
// TDM: one descriptor-driven DMA for the VT tile (2D: 32 keys x 128 dims,
// row stride Sc elements). D# per ISA 8.3/8.4; groups 2/3 unused (<=2D).
DEVINL void tdm_load_vt(const __bf16* __restrict__ gsrc, uint32_t lds_off) {
  const uint64_t ga = (uint64_t)(uintptr_t)gsrc;
  u32x4 g0;
  g0[0] = 1u;                                   // count=1, user descriptor
  g0[1] = lds_off;                              // lds_addr (bytes)
  g0[2] = (uint32_t)ga;                         // global_addr[31:0]
  g0[3] = (uint32_t)((ga >> 32) & 0x01FFFFFFu)  // global_addr[56:32]
          | 0x80000000u;                        // type = 2 ("image")
  i32x8 g1;
  g1[0] = 1 << 16;                              // data_size = 1 (2 bytes)
  g1[1] = (Sc & 0xFFFF) << 16;                  // tensor_dim0[15:0]
  g1[2] = ((Sc >> 16) & 0xFFFF) | (Dc << 16);   // tensor_dim0[31:16] | tensor_dim1[15:0]
  g1[3] = KT << 16;                             // tile_dim0 = 32
  g1[4] = Dc;                                   // tile_dim1 = 128
  g1[5] = Sc;                                   // tensor_dim0_stride[31:0]
  g1[6] = 0;
  g1[7] = 0;
  const i32x4 z4 = {0, 0, 0, 0};
#if __clang_major__ >= 23
  const i32x8 z8 = {0, 0, 0, 0, 0, 0, 0, 0};
  __builtin_amdgcn_tensor_load_to_lds(g0, g1, z4, z4, z8, 0);
#else
  __builtin_amdgcn_tensor_load_to_lds(g0, g1, z4, z4, 0);
#endif
}
#else
// fallback: per-lane async copy of the VT tile (rows strided by S)
DEVINL void async_vt_tile(const __bf16* __restrict__ grow0, __bf16* ldst, int tid) {
  const __bf16* g = grow0 + (size_t)tid * Sc;
  __bf16* l = ldst + tid * KT;
  #pragma unroll
  for (int i = 0; i < 4; ++i) {
    asm volatile("global_load_async_to_lds_b128 %0, %1, off"
                 :: "v"((uint32_t)(uintptr_t)(l + i * 8)),
                    "v"((uint64_t)(uintptr_t)(g + i * 8)) : "memory");
  }
}
#endif

// ---------------- flash-attention core ----------------

template <bool MASK>
DEVINL void attn_tile(const __bf16* __restrict__ gqrow, // global bf16 Q row (qbase+ln)
                      const __bf16* __restrict__ sK,    // KT x Dc row-major bf16
                      const __bf16* __restrict__ sVT,   // Dc x KT bf16
                      __bf16* __restrict__ sPw,         // 16 x KT per-wave bounce
                      v8f (&o)[8], float (&m)[8], float (&l)[8],
                      int qbase, int j0, int ln, int hi)
{
  AFrag qf[4];
  #pragma unroll
  for (int kb4 = 0; kb4 < 4; ++kb4) {
    const int kb = kb4 * 32;
    qf[kb4].h[0] = *(const v8bf*)(gqrow + kb + hi * 16);
    qf[kb4].h[1] = *(const v8bf*)(gqrow + kb + hi * 16 + 8);
  }

  v8f accA = {};
  v8f accB = {};
  #pragma unroll
  for (int kb4 = 0; kb4 < 4; ++kb4) {
    const int kb = kb4 * 32;
    v16bf bA = *(const v16bf*)(sK + ln * Dc        + kb + hi * 16);
    v16bf bB = *(const v16bf*)(sK + (16 + ln) * Dc + kb + hi * 16);
    accA = wmma_bf16(qf[kb4].v, bA, accA);
    accB = wmma_bf16(qf[kb4].v, bB, accB);
  }

  #pragma unroll
  for (int r = 0; r < 8; ++r) {
    float sa = accA[r] * SCALE_LOG2;
    float sb = accB[r] * SCALE_LOG2;
    if (MASK) {
      const int row = qbase + hi * 8 + r;
      if (j0 + ln > row)      sa = -1e30f;
      if (j0 + 16 + ln > row) sb = -1e30f;
    }
    const float mx = half_max(fmaxf(sa, sb));
    const float mnew = fmaxf(m[r], mx);
    const float corr = EXP2F(m[r] - mnew);
    const float pa = EXP2F(sa - mnew);
    const float pb = EXP2F(sb - mnew);
    const float rs = half_sum(pa + pb);
    l[r] = l[r] * corr + rs;
    m[r] = mnew;
    #pragma unroll
    for (int nt = 0; nt < 8; ++nt) o[nt][r] *= corr;
    sPw[(hi * 8 + r) * KT + ln]      = (__bf16)pa;
    sPw[(hi * 8 + r) * KT + 16 + ln] = (__bf16)pb;
  }

  AFrag p;
  const __bf16* pp = sPw + ln * KT;
  p.h[0] = *(const v8bf*)(pp + hi * 16);
  p.h[1] = *(const v8bf*)(pp + hi * 16 + 8);
  #pragma unroll
  for (int nt = 0; nt < 8; ++nt) {
    v16bf bv = *(const v16bf*)(sVT + (nt * 16 + ln) * KT + hi * 16);
    o[nt] = wmma_bf16(p.v, bv, o[nt]);
  }
}

__global__ __launch_bounds__(128) void diffattn_fa_kernel(
    const __bf16* __restrict__ q1b, const __bf16* __restrict__ k1b,
    const __bf16* __restrict__ q2b, const __bf16* __restrict__ k2b,
    const __bf16* __restrict__ vtb, const float* __restrict__ glam,
    float* __restrict__ gout)
{
  __shared__ __align__(64) __bf16 sK1[2][KT * Dc];   // 16 KB
  __shared__ __align__(64) __bf16 sK2[2][KT * Dc];   // 16 KB
  __shared__ __align__(64) __bf16 sVT[2][Dc * KT];   // 16 KB
  __shared__ __align__(64) __bf16 sP[4][16 * KT];    //  4 KB

  const int tid  = threadIdx.x;
  const int wave = tid >> 5;
  const int lane = tid & 31;
  const int ln   = lane & 15;
  const int hi   = lane >> 4;

  const int qt = blockIdx.x & 31;
  const int bh = blockIdx.x >> 5;
  const int q0 = qt * QT;
  const int qbase = q0 + wave * 16;
  const size_t base = (size_t)bh * Sc * Dc;
  const float lam = __expf(glam[0]);

  const __bf16* q1row = q1b + base + (size_t)(qbase + ln) * Dc;
  const __bf16* q2row = q2b + base + (size_t)(qbase + ln) * Dc;

  v8f o1[8], o2[8];
  float m1[8], l1[8], m2[8], l2[8];
  #pragma unroll
  for (int i = 0; i < 8; ++i) {
    o1[i] = {}; o2[i] = {};
    m1[i] = -1e30f; l1[i] = 0.f;
    m2[i] = -1e30f; l2[i] = 0.f;
  }

  const __bf16* vt0 = vtb + (size_t)bh * Dc * Sc;  // row d stride = Sc

  // prime the pipeline for tile 0
  async_tile_8k(k1b + base, &sK1[0][0], tid);
  async_tile_8k(k2b + base, &sK2[0][0], tid);
#if HAVE_TDM
  if (wave == 0) tdm_load_vt(vt0, (uint32_t)(uintptr_t)&sVT[0][0]);
#else
  async_vt_tile(vt0, &sVT[0][0], tid);
#endif

  const int jend = q0 + QT;
  for (int j0 = 0; j0 < jend; j0 += KT) {
    const int  cur = (j0 / KT) & 1;
    const bool have_next = (j0 + KT) < jend;

    __syncthreads();  // all waves done computing on buffer cur^1

    if (have_next) {
      const size_t nb = base + (size_t)(j0 + KT) * Dc;
      async_tile_8k(k1b + nb, &sK1[cur ^ 1][0], tid);
      async_tile_8k(k2b + nb, &sK2[cur ^ 1][0], tid);
#if HAVE_TDM
      asm volatile("s_wait_asynccnt 0x8" ::: "memory");  // retire this tile's K asyncs
#else
      async_vt_tile(vt0 + j0 + KT, &sVT[cur ^ 1][0], tid);
      asm volatile("s_wait_asynccnt 0xc" ::: "memory");
#endif
    } else {
      asm volatile("s_wait_asynccnt 0x0" ::: "memory");
    }

#if HAVE_TDM
    if (wave == 0) {   // wave 0 owns the TDM pipe; barrier releases the others
      if (have_next) {
        tdm_load_vt(vt0 + j0 + KT, (uint32_t)(uintptr_t)&sVT[cur ^ 1][0]);
        __builtin_amdgcn_s_wait_tensorcnt(1);
      } else {
        __builtin_amdgcn_s_wait_tensorcnt(0);
      }
    }
#endif

    __syncthreads();  // current tile fully resident for all waves

    if (j0 + KT - 1 <= qbase) {           // below the diagonal: no mask
      attn_tile<false>(q1row, sK1[cur], sVT[cur], sP[wave], o1, m1, l1, qbase, j0, ln, hi);
      attn_tile<false>(q2row, sK2[cur], sVT[cur], sP[wave], o2, m2, l2, qbase, j0, ln, hi);
    } else if (j0 <= qbase + 15) {        // diagonal tile: causal mask
      attn_tile<true>(q1row, sK1[cur], sVT[cur], sP[wave], o1, m1, l1, qbase, j0, ln, hi);
      attn_tile<true>(q2row, sK2[cur], sVT[cur], sP[wave], o2, m2, l2, qbase, j0, ln, hi);
    }
  }

  // epilogue: out = O1/l1 - lambda * O2/l2
  #pragma unroll
  for (int r = 0; r < 8; ++r) {
    const float inv1 = 1.0f / l1[r];
    const float inv2 = lam / l2[r];
    const int row = qbase + hi * 8 + r;
    float* orow = gout + base + (size_t)row * Dc;
    #pragma unroll
    for (int nt = 0; nt < 8; ++nt) {
      orow[nt * 16 + ln] = o1[nt][r] * inv1 - o2[nt][r] * inv2;
    }
  }
}

// ---------------- host launcher ----------------

extern "C" void kernel_launch(void* const* d_in, const int* in_sizes, int n_in,
                              void* d_out, int out_size, void* d_ws, size_t ws_size,
                              hipStream_t stream) {
  (void)in_sizes; (void)n_in; (void)out_size; (void)ws_size;
  const float* q1  = (const float*)d_in[0];
  const float* k1  = (const float*)d_in[1];
  const float* v   = (const float*)d_in[2];
  const float* q2  = (const float*)d_in[3];
  const float* k2  = (const float*)d_in[4];
  const float* lam = (const float*)d_in[5];
  float* out = (float*)d_out;

  // workspace layout: 5 bf16 tensors of B*H*S*D elements (~63 MB)
  const size_t EL = (size_t)Bc * Hc * Sc * Dc;   // 6,291,456
  __bf16* ws  = (__bf16*)d_ws;
  __bf16* q1b = ws;
  __bf16* k1b = ws + EL;
  __bf16* q2b = ws + 2 * EL;
  __bf16* k2b = ws + 3 * EL;
  __bf16* vtb = ws + 4 * EL;

  const int cvt_blocks = (int)(EL / (256 * 4));  // 6144, exact
  cvt_bf16_kernel<<<cvt_blocks, 256, 0, stream>>>(q1, q1b);
  cvt_bf16_kernel<<<cvt_blocks, 256, 0, stream>>>(k1, k1b);
  cvt_bf16_kernel<<<cvt_blocks, 256, 0, stream>>>(q2, q2b);
  cvt_bf16_kernel<<<cvt_blocks, 256, 0, stream>>>(k2, k2b);
  vT_bf16_kernel<<<Bc * Hc * (Sc / 32) * (Dc / 32), 256, 0, stream>>>(v, vtb);

  diffattn_fa_kernel<<<Bc * Hc * (Sc / QT), 128, 0, stream>>>(
      q1b, k1b, q2b, k2b, vtb, lam, out);
}
